// SpikingSpatioTemporalAttention_63995012711047
// MI455X (gfx1250) — matrix operand order repaired and checked by reading, not compiled, not run on hardware
//
#include <hip/hip_runtime.h>

#define T_ 64
#define B_ 32
#define C_ 256
#define V_ 25
#define H_ 8
#define HD_ 32
#define P_ (T_ * B_ * V_)          // 51200 positions
#define NTOT_ (T_ * B_ * C_ * V_)  // 13107200 elements per full tensor
#define NS_ (B_ * C_ * V_)         // 204800 = per-timestep elements = T-stride
#define EPS_ 1e-5f

typedef __attribute__((ext_vector_type(16))) __bf16 v16bf;
typedef __attribute__((ext_vector_type(8)))  __bf16 v8bf;
typedef __attribute__((ext_vector_type(8)))  float  v8f;

// float -> bf16 with round-to-nearest-even
static __device__ __forceinline__ __bf16 f2bf(float f) {
  unsigned u = __builtin_bit_cast(unsigned, f);
  unsigned r = u + 0x7FFFu + ((u >> 16) & 1u);
  return __builtin_bit_cast(__bf16, (unsigned short)(r >> 16));
}

// ---------------------------------------------------------------------------
// Prep: fold BN (and proj bias) into bf16 weights + fp32 bias.
//   q/k/v : y = conv*s + (beta - mean*s),              s = gamma*rsqrt(var+eps)
//   proj  : y = conv*s + ((proj_b - mean)*s + beta)
// ---------------------------------------------------------------------------
__global__ void prep_weights(
    const float* qw, const float* kw, const float* vw, const float* pw,
    const float* projb,
    const float* qg, const float* qbe, const float* qm, const float* qva,
    const float* kg, const float* kbe, const float* km, const float* kva,
    const float* vg, const float* vbe, const float* vm, const float* vva,
    const float* pg, const float* pbe, const float* pm, const float* pva,
    __bf16* wbf, float* bias) {
  int mat = blockIdx.x;   // 0=q 1=k 2=v 3=proj
  int o   = threadIdx.x;  // 0..255 output channel
  const float *W, *g, *be, *m, *va;
  if (mat == 0)      { W = qw; g = qg; be = qbe; m = qm; va = qva; }
  else if (mat == 1) { W = kw; g = kg; be = kbe; m = km; va = kva; }
  else if (mat == 2) { W = vw; g = vg; be = vbe; m = vm; va = vva; }
  else               { W = pw; g = pg; be = pbe; m = pm; va = pva; }
  float s  = g[o] * rsqrtf(va[o] + EPS_);
  float bi = (mat == 3) ? (projb[o] - m[o]) * s + be[o] : be[o] - m[o] * s;
  bias[mat * C_ + o] = bi;
  __bf16* dst      = wbf + (size_t)mat * C_ * C_ + (size_t)o * C_;
  const float* src = W + (size_t)o * C_;
  for (int c = 0; c < C_; ++c) dst[c] = f2bf(src[c] * s);
}

// ---------------------------------------------------------------------------
// Prep inputs: qkv_in = a*|x_t - x_{t-1}| + (1-a)*x ; pack x and qkv_in to
// bf16 in position-major (P, C) layout for the WMMA GEMMs.
// ---------------------------------------------------------------------------
__global__ void prep_inputs(const float* __restrict__ x,
                            const float* __restrict__ alpha,
                            __bf16* __restrict__ xbf, __bf16* __restrict__ gbf) {
  size_t tid = (size_t)blockIdx.x * blockDim.x + threadIdx.x;  // < NTOT_
  int t  = (int)(tid / NS_);
  int r  = (int)(tid % NS_);        // (b,c,vv)
  int b  = r / (C_ * V_);
  int r2 = r % (C_ * V_);
  int c  = r2 / V_;
  int vv = r2 % V_;
  float cur  = x[tid];
  float prev = t ? x[tid - NS_] : cur;
  float a    = alpha[c];
  float gv   = a * fabsf(cur - prev) + (1.f - a) * cur;
  size_t p   = (size_t)(t * B_ + b) * V_ + vv;
  xbf[p * C_ + c] = f2bf(cur);
  gbf[p * C_ + c] = f2bf(gv);
}

// softmax(base + 0.5*learned) over last dim, per head
__global__ void prep_topo(const float* __restrict__ base,
                          const float* __restrict__ learned,
                          float* __restrict__ topo) {
  int tid = blockIdx.x * blockDim.x + threadIdx.x;
  if (tid >= H_ * V_) return;
  int h = tid / V_;
  int i = tid % V_;
  float row[V_];
  float mx = -1e30f;
  for (int j = 0; j < V_; ++j) {
    float l = base[i * V_ + j] + 0.5f * learned[(h * V_ + i) * V_ + j];
    row[j] = l; mx = fmaxf(mx, l);
  }
  float sum = 0.f;
  for (int j = 0; j < V_; ++j) { row[j] = __expf(row[j] - mx); sum += row[j]; }
  float inv = 1.f / sum;
  for (int j = 0; j < V_; ++j) topo[(h * V_ + i) * V_ + j] = row[j] * inv;
}

// ---------------------------------------------------------------------------
// WMMA GEMM, register-blocked: out(P,C) = in(P,C) x W'(C,C)^T + bias.
// Each wave computes a 16-position x 64-oc strip: one A fragment (16x32)
// is reused across 4 B tiles (32x16) -> 4 accumulators, A traffic / 4.
// Block = 128 threads (4 waves) covers all 256 output channels.
// Operand register layouts used here (same k order for A and B):
//   A 16x32 bf16 : lane m=lane&15, elems 0..15 = K[kh*16 .. kh*16+15]
//   B 32x16 bf16 : lane n=lane&15, elems 0..15 = K[kh*16 .. kh*16+15]
//   D f32 16x16  : VGPR g -> (M = kh*8+g, N = lane&15)
// ---------------------------------------------------------------------------
__global__ void __launch_bounds__(128) gemm_bf16(
    const __bf16* __restrict__ in, const __bf16* __restrict__ w,
    const float* __restrict__ bias, float* __restrict__ out) {
  int lane  = threadIdx.x & 31;
  int wave  = threadIdx.x >> 5;   // 0..3 -> oc block of 64
  int lm    = lane & 15;
  int kh    = lane >> 4;
  int ptile = blockIdx.x;
  int ocb   = wave * 64;

  const __bf16* arow = in + (size_t)(ptile * 16 + lm) * C_;
  const __bf16* brow = w  + (size_t)(ocb + lm) * C_;   // B tile i at +i*16*C_

  v8f acc0 = {0.f,0.f,0.f,0.f,0.f,0.f,0.f,0.f};
  v8f acc1 = acc0, acc2 = acc0, acc3 = acc0;

#pragma unroll
  for (int k = 0; k < 8; ++k) {
    int c0 = k * 32;
    __builtin_prefetch((const void*)(arow + c0 + 64), 0, 1);  // global_prefetch_b8
    v8bf alo = *(const v8bf*)(arow + c0 + kh * 16);
    v8bf ahi = *(const v8bf*)(arow + c0 + kh * 16 + 8);
    v16bf Am = __builtin_shufflevector(alo, ahi, 0,1,2,3,4,5,6,7,8,9,10,11,12,13,14,15);

    const __bf16* bp0 = brow + c0 + kh * 16;
    v8bf b0l = *(const v8bf*)(bp0);
    v8bf b0h = *(const v8bf*)(bp0 + 8);
    v8bf b1l = *(const v8bf*)(bp0 + 16 * C_);
    v8bf b1h = *(const v8bf*)(bp0 + 16 * C_ + 8);
    v8bf b2l = *(const v8bf*)(bp0 + 32 * C_);
    v8bf b2h = *(const v8bf*)(bp0 + 32 * C_ + 8);
    v8bf b3l = *(const v8bf*)(bp0 + 48 * C_);
    v8bf b3h = *(const v8bf*)(bp0 + 48 * C_ + 8);

    v16bf B0 = __builtin_shufflevector(b0l, b0h, 0,1,2,3,4,5,6,7,8,9,10,11,12,13,14,15);
    v16bf B1 = __builtin_shufflevector(b1l, b1h, 0,1,2,3,4,5,6,7,8,9,10,11,12,13,14,15);
    v16bf B2 = __builtin_shufflevector(b2l, b2h, 0,1,2,3,4,5,6,7,8,9,10,11,12,13,14,15);
    v16bf B3 = __builtin_shufflevector(b3l, b3h, 0,1,2,3,4,5,6,7,8,9,10,11,12,13,14,15);

    acc0 = __builtin_amdgcn_wmma_f32_16x16x32_bf16(false, Am, false, B0, (short)0, acc0, false, false);
    acc1 = __builtin_amdgcn_wmma_f32_16x16x32_bf16(false, Am, false, B1, (short)0, acc1, false, false);
    acc2 = __builtin_amdgcn_wmma_f32_16x16x32_bf16(false, Am, false, B2, (short)0, acc2, false, false);
    acc3 = __builtin_amdgcn_wmma_f32_16x16x32_bf16(false, Am, false, B3, (short)0, acc3, false, false);
  }

  int mb = kh * 8;
  float b0 = bias[ocb + lm];
  float b1 = bias[ocb + 16 + lm];
  float b2 = bias[ocb + 32 + lm];
  float b3 = bias[ocb + 48 + lm];
  float* obase = out + (size_t)(ptile * 16 + mb) * C_ + ocb + lm;
#pragma unroll
  for (int g = 0; g < 8; ++g) {
    float* orow = obase + (size_t)g * C_;
    orow[0]  = acc0[g] + b0;
    orow[16] = acc1[g] + b1;
    orow[32] = acc2[g] + b2;
    orow[48] = acc3[g] + b3;
  }
}

// ---------------------------------------------------------------------------
// Multi-step LIF (tau=2, hard reset) in place over the T axis.
// Layout (P,C): element (t,b,c,v) at t*NS_ + (b*V+v)*C + c  -> base = tid.
// ---------------------------------------------------------------------------
__global__ void lif_kernel(float* buf, float th) {
  int tid = blockIdx.x * blockDim.x + threadIdx.x;  // < NS_
  float v = 0.f;
  size_t idx = (size_t)tid;
  for (int t = 0; t < T_; ++t) {
    float vn = 0.5f * (v + buf[idx]);     // v += (x - v)/tau, tau = 2
    float s  = (vn >= th) ? 1.f : 0.f;
    buf[idx] = s;
    v = vn * (1.f - s);                   // hard reset
    idx += NS_;
  }
}

// kv_local = k_s * v_s ; kv_spatial[i] = sum_j topo[h,i,j] * kv_local[j]
// One thread owns a full V-slice for fixed (t,b,c) -> safe in-place into vs.
__global__ void topo_mix(const float* __restrict__ ks, const float* vs,
                         const float* __restrict__ topo, float* out) {
  int tid = blockIdx.x * blockDim.x + threadIdx.x;  // < T_*B_*C_
  int c  = tid % C_;
  int tb = tid / C_;
  int h  = c / HD_;
  size_t pbase = (size_t)tb * V_;
  float kv[V_];
#pragma unroll
  for (int j = 0; j < V_; ++j) {
    size_t idx = (pbase + j) * C_ + c;
    kv[j] = ks[idx] * vs[idx];
  }
  const float* tp = topo + h * V_ * V_;
  for (int i = 0; i < V_; ++i) {
    float a = 0.f;
#pragma unroll
    for (int j = 0; j < V_; ++j) a += tp[i * V_ + j] * kv[j];
    out[(pbase + i) * C_ + c] = a;
  }
}

// S_t = d*S_{t-1} + (1-d)*kv_t ; out_t = q_t * S_t ; d per channel (h,hd)=c.
// Emits bf16 directly into the proj-GEMM input buffer (saves a pack pass).
__global__ void ssre_kernel(const float* __restrict__ qs,
                            const float* __restrict__ kvs,
                            const float* __restrict__ sdw,
                            __bf16* __restrict__ out) {
  int tid = blockIdx.x * blockDim.x + threadIdx.x;  // < NS_
  float w   = sdw[tid % C_];                        // (H,1,HD) flat == channel
  float dec = 1.f / (1.f + __expf(-w));
  dec = fminf(fmaxf(dec, 0.01f), 0.99f);
  float S = 0.f;
  size_t idx = (size_t)tid;
  for (int t = 0; t < T_; ++t) {
    S = dec * S + (1.f - dec) * kvs[idx];
    out[idx] = f2bf(qs[idx] * S);
    idx += NS_;
  }
}

// Final LIF(th=0.5) on proj pre-activation + identity add, writing (T,B,C,V).
__global__ void final_kernel(const float* __restrict__ pre,
                             const float* __restrict__ x,
                             float* __restrict__ out) {
  int tid = blockIdx.x * blockDim.x + threadIdx.x;  // < NS_, enumerates (b,c,vv)
  int b  = tid / (C_ * V_);
  int r  = tid % (C_ * V_);
  int c  = r / V_;
  int vv = r % V_;
  size_t pidx = (size_t)(b * V_ + vv) * C_ + c;       // (P,C) layout
  size_t xidx = (size_t)(b * C_ + c) * V_ + vv;       // (T,B,C,V) layout
  float v = 0.f;
  for (int t = 0; t < T_; ++t) {
    float vn = 0.5f * (v + pre[pidx]);
    float s  = (vn >= 0.5f) ? 1.f : 0.f;
    out[xidx] = s + x[xidx];
    v = vn * (1.f - s);
    pidx += NS_;
    xidx += NS_;
  }
}

// ---------------------------------------------------------------------------
extern "C" void kernel_launch(void* const* d_in, const int* in_sizes, int n_in,
                              void* d_out, int out_size, void* d_ws, size_t ws_size,
                              hipStream_t stream) {
  (void)in_sizes; (void)n_in; (void)out_size; (void)ws_size;
  const float* x       = (const float*)d_in[0];
  const float* alpha   = (const float*)d_in[1];
  const float* qw      = (const float*)d_in[2];
  const float* kw      = (const float*)d_in[3];
  const float* vw      = (const float*)d_in[4];
  const float* basetp  = (const float*)d_in[5];
  const float* learned = (const float*)d_in[6];
  const float* sdw     = (const float*)d_in[7];
  const float* pw      = (const float*)d_in[8];
  const float* pb      = (const float*)d_in[9];
  const float *qg = (const float*)d_in[10], *qbe = (const float*)d_in[11],
              *qm = (const float*)d_in[12], *qva = (const float*)d_in[13];
  const float *kg = (const float*)d_in[14], *kbe = (const float*)d_in[15],
              *km = (const float*)d_in[16], *kva = (const float*)d_in[17];
  const float *vg = (const float*)d_in[18], *vbe = (const float*)d_in[19],
              *vm = (const float*)d_in[20], *vva = (const float*)d_in[21];
  const float *pg = (const float*)d_in[22], *pbe = (const float*)d_in[23],
              *pm = (const float*)d_in[24], *pva = (const float*)d_in[25];

  char* ws = (char*)d_ws;
  size_t off = 0;
  float*  bufA = (float*)(ws + off);  off += (size_t)NTOT_ * 4;
  float*  bufB = (float*)(ws + off);  off += (size_t)NTOT_ * 4;
  float*  bufC = (float*)(ws + off);  off += (size_t)NTOT_ * 4;
  __bf16* xbf  = (__bf16*)(ws + off); off += (size_t)NTOT_ * 2;
  __bf16* gbf  = (__bf16*)(ws + off); off += (size_t)NTOT_ * 2;
  __bf16* wbf  = (__bf16*)(ws + off); off += (size_t)4 * C_ * C_ * 2;
  float*  bias = (float*)(ws + off);  off += (size_t)4 * C_ * 4;
  float*  topo = (float*)(ws + off);  off += (size_t)H_ * V_ * V_ * 4;

  prep_weights<<<4, 256, 0, stream>>>(qw, kw, vw, pw, pb,
                                      qg, qbe, qm, qva, kg, kbe, km, kva,
                                      vg, vbe, vm, vva, pg, pbe, pm, pva,
                                      wbf, bias);
  prep_inputs<<<NTOT_ / 256, 256, 0, stream>>>(x, alpha, xbf, gbf);
  prep_topo<<<1, 256, 0, stream>>>(basetp, learned, topo);

  gemm_bf16<<<P_ / 16, 128, 0, stream>>>(xbf, wbf + 2 * C_ * C_, bias + 2 * C_, bufA); // V
  gemm_bf16<<<P_ / 16, 128, 0, stream>>>(gbf, wbf + 0 * C_ * C_, bias + 0 * C_, bufB); // Q
  gemm_bf16<<<P_ / 16, 128, 0, stream>>>(gbf, wbf + 1 * C_ * C_, bias + 1 * C_, bufC); // K

  lif_kernel<<<NS_ / 256, 256, 0, stream>>>(bufA, 1.0f);  // v_s
  lif_kernel<<<NS_ / 256, 256, 0, stream>>>(bufB, 1.0f);  // q_s
  lif_kernel<<<NS_ / 256, 256, 0, stream>>>(bufC, 1.0f);  // k_s

  topo_mix<<<(T_ * B_ * C_) / 256, 256, 0, stream>>>(bufC, bufA, topo, bufA);
  lif_kernel<<<NS_ / 256, 256, 0, stream>>>(bufA, 0.5f);                // kv_spike
  ssre_kernel<<<NS_ / 256, 256, 0, stream>>>(bufB, bufA, sdw, gbf);     // mem -> bf16
  gemm_bf16<<<P_ / 16, 128, 0, stream>>>(gbf, wbf + 3 * C_ * C_, bias + 3 * C_, bufA); // proj
  final_kernel<<<NS_ / 256, 256, 0, stream>>>(bufA, x, (float*)d_out);
}
